// TFWav2Vec2Attention_19000935317958
// MI455X (gfx1250) — matrix operand order repaired and checked by reading, not compiled, not run on hardware
//
#include <hip/hip_runtime.h>
#include <hip/hip_bf16.h>

typedef __attribute__((ext_vector_type(16))) _Float16 v16h;
typedef __attribute__((ext_vector_type(8)))  _Float16 h8;
typedef __attribute__((ext_vector_type(8)))  float    v8f;
typedef int i4 __attribute__((vector_size(16)));  // matches builtin's V4i

#define T_SEQ   2048
#define D_MODEL 1024
#define N_HEADS 16
#define HEAD_DIM 64
#define N_BATCH 2

// ---------------------------------------------------------------------------
// Async global->LDS staging (CDNA5 GLOBAL_LOAD_ASYNC_TO_LDS_B128, ASYNCcnt),
// with a synchronous fallback if the builtin is unavailable on this toolchain.
// Builtin signature (from compiler diagnostic): (v4i __device__* src,
// v4i __shared__* dst, imm int offset, imm int cpol).
// ---------------------------------------------------------------------------
#if __has_builtin(__builtin_amdgcn_global_load_async_to_lds_b128) && \
    __has_builtin(__builtin_amdgcn_s_wait_asynccnt)
#define HAS_ASYNC_LDS 1
#else
#define HAS_ASYNC_LDS 0
#endif

__device__ __forceinline__ void copy16_async(const _Float16* gsrc, _Float16* lds) {
#if HAS_ASYNC_LDS
  __attribute__((address_space(1))) i4* g =
      (__attribute__((address_space(1))) i4*)(uintptr_t)gsrc;
  __attribute__((address_space(3))) i4* l =
      (__attribute__((address_space(3))) i4*)(unsigned)(uintptr_t)lds;
  __builtin_amdgcn_global_load_async_to_lds_b128(g, l, 0, 0);
#else
  *(h8*)lds = *(const h8*)gsrc;
#endif
}

__device__ __forceinline__ void async_wait0() {
#if HAS_ASYNC_LDS
  __builtin_amdgcn_s_wait_asynccnt(0);
#endif
}

// ---------------------------------------------------------------------------
// WMMA fragment helpers, wave32, A/B in contiguous k order (not ISA table order); C/D per ISA.
// A (16x32 f16): lane L holds row M=L&15; element i -> K = 16*hi + i.
// B (32x16 f16): lane L holds col N=L&15; element i -> K = 16*hi + i.
// C/D (16x16 f32): element r of lane L -> M = r + 8*(L>>4), N = L&15.
// ---------------------------------------------------------------------------
__device__ __forceinline__ v16h frag_a(const _Float16* p, int ld) {
  const int lane = threadIdx.x & 31;
  const int row  = lane & 15;
  const int hi   = lane >> 4;
  v16h a;
#pragma unroll
  for (int i = 0; i < 16; ++i) {
    const int k = 16 * hi + i;
    a[i] = p[row * ld + k];
  }
  return a;
}

__device__ __forceinline__ v16h frag_b(const _Float16* p, int ld) {
  const int lane = threadIdx.x & 31;
  const int col  = lane & 15;
  const int hi   = lane >> 4;
  v16h b;
#pragma unroll
  for (int i = 0; i < 16; ++i) {
    const int k = 16 * hi + i;
    b[i] = p[k * ld + col];
  }
  return b;
}

__device__ __forceinline__ v16h frag_bT(const _Float16* p, int ld) {
  // B[k][n] = p[n*ld + k]  (transposed source, used for Q . K^T)
  const int lane = threadIdx.x & 31;
  const int col  = lane & 15;
  const int hi   = lane >> 4;
  v16h b;
#pragma unroll
  for (int i = 0; i < 16; ++i) {
    const int k = 16 * hi + i;
    b[i] = p[col * ld + k];
  }
  return b;
}

__device__ __forceinline__ v8f wmma32(v16h a, v16h b, v8f c) {
  return __builtin_amdgcn_wmma_f32_16x16x32_f16(false, a, false, b, (short)0, c,
                                                false, false);
}

// ---------------------------------------------------------------------------
// Stage 1: fp32 -> f16 convert
// ---------------------------------------------------------------------------
__global__ __launch_bounds__(256) void cvt_f32_to_f16(
    const float* __restrict__ src, _Float16* __restrict__ dst, int n) {
  int i = (blockIdx.x * 256 + threadIdx.x) * 4;
  if (i + 3 < n) {
    float4 v = *(const float4*)&src[i];
    dst[i]     = (_Float16)v.x;
    dst[i + 1] = (_Float16)v.y;
    dst[i + 2] = (_Float16)v.z;
    dst[i + 3] = (_Float16)v.w;
  }
}

// ---------------------------------------------------------------------------
// Stage 2/6: GEMM  out = (A[MxK] * W[KxN] + bias) * scale
// 256 thr (8 waves), block tile 64x128, K-step 32, 2x2 subtiles per wave,
// double-buffered async LDS staging.
// ---------------------------------------------------------------------------
template <bool OUT_F32>
__global__ __launch_bounds__(256) void gemm_bias(
    const _Float16* __restrict__ A, const _Float16* __restrict__ W,
    const float* __restrict__ bias, float scale, _Float16* __restrict__ outH,
    float* __restrict__ outF, int M, int N, int K) {
  __shared__ _Float16 As[2][64][40];   // 64x32 + pad, double buffered
  __shared__ _Float16 Bs[2][32][136];  // 32x128 + pad, double buffered
  const int tid  = threadIdx.x;
  const int wave = tid >> 5;
  const int lane = tid & 31;
  const int hi   = lane >> 4;
  const int col  = lane & 15;
  const int bm   = blockIdx.x * 64;
  const int bn   = blockIdx.y * 128;
  const int mB   = (wave & 1) * 2;   // m-subtile pair base (0 or 2)
  const int nB   = (wave >> 1) * 2;  // n-subtile pair base (0,2,4,6)

  const int ar = tid >> 2, ac = (tid & 3) * 8;   // A loader: 64x32, 8 h/thr
  const int br = tid >> 3, bc = (tid & 7) * 16;  // B loader: 32x128, 16 h/thr

  v8f acc00 = {}, acc01 = {}, acc10 = {}, acc11 = {};

  // prologue: stage tile 0 into buffer 0
  copy16_async(&A[(size_t)(bm + ar) * K + ac], &As[0][ar][ac]);
  copy16_async(&W[(size_t)br * N + bn + bc], &Bs[0][br][bc]);
  copy16_async(&W[(size_t)br * N + bn + bc + 8], &Bs[0][br][bc + 8]);

  for (int k0 = 0; k0 < K; k0 += 32) {
    const int buf = (k0 >> 5) & 1;
    async_wait0();
    __syncthreads();  // buffer `buf` ready for everyone
    if (k0 + 32 < K) {  // overlap: stage next tile while computing
      copy16_async(&A[(size_t)(bm + ar) * K + k0 + 32 + ac], &As[buf ^ 1][ar][ac]);
      copy16_async(&W[(size_t)(k0 + 32 + br) * N + bn + bc], &Bs[buf ^ 1][br][bc]);
      copy16_async(&W[(size_t)(k0 + 32 + br) * N + bn + bc + 8],
                   &Bs[buf ^ 1][br][bc + 8]);
    }
    v16h a0 = frag_a(&As[buf][(mB + 0) * 16][0], 40);
    v16h a1 = frag_a(&As[buf][(mB + 1) * 16][0], 40);
    v16h b0 = frag_b(&Bs[buf][0][(nB + 0) * 16], 136);
    v16h b1 = frag_b(&Bs[buf][0][(nB + 1) * 16], 136);
    acc00 = wmma32(a0, b0, acc00);
    acc01 = wmma32(a0, b1, acc01);
    acc10 = wmma32(a1, b0, acc10);
    acc11 = wmma32(a1, b1, acc11);
    __syncthreads();  // everyone done reading `buf` before it is restaged
  }

  const int gc0 = bn + (nB + 0) * 16 + col;
  const int gc1 = bn + (nB + 1) * 16 + col;
  const float bi0 = bias[gc0];
  const float bi1 = bias[gc1];
#pragma unroll
  for (int r = 0; r < 8; ++r) {
    const int gr0 = bm + (mB + 0) * 16 + r + 8 * hi;
    const int gr1 = bm + (mB + 1) * 16 + r + 8 * hi;
    float v00 = (acc00[r] + bi0) * scale;
    float v01 = (acc01[r] + bi1) * scale;
    float v10 = (acc10[r] + bi0) * scale;
    float v11 = (acc11[r] + bi1) * scale;
    if constexpr (OUT_F32) {
      outF[(size_t)gr0 * N + gc0] = v00;
      outF[(size_t)gr0 * N + gc1] = v01;
      outF[(size_t)gr1 * N + gc0] = v10;
      outF[(size_t)gr1 * N + gc1] = v11;
    } else {
      outH[(size_t)gr0 * N + gc0] = (_Float16)v00;
      outH[(size_t)gr0 * N + gc1] = (_Float16)v01;
      outH[(size_t)gr1 * N + gc0] = (_Float16)v10;
      outH[(size_t)gr1 * N + gc1] = (_Float16)v11;
    }
  }
}

// ---------------------------------------------------------------------------
// Stage 3: scores = Q . K^T + mask  -> attn_weights logits.
// grid: (T/64 qtiles, T/128 ktiles, B*H); 2x2 subtiles per wave.
// ---------------------------------------------------------------------------
__global__ __launch_bounds__(256) void scores_kernel(
    const _Float16* __restrict__ Q, const _Float16* __restrict__ Kc,
    const float* __restrict__ mask, float* __restrict__ attnW) {
  __shared__ _Float16 Qs[64][72];
  __shared__ _Float16 Ks[128][72];
  const int tid  = threadIdx.x;
  const int wave = tid >> 5, lane = tid & 31;
  const int hi = lane >> 4, col = lane & 15;
  const int bh = blockIdx.z, b = bh >> 4, h = bh & 15;
  const int q0 = blockIdx.x * 64, p0 = blockIdx.y * 128;
  const _Float16* Qp = Q + (size_t)b * T_SEQ * D_MODEL + h * HEAD_DIM;
  const _Float16* Kp = Kc + (size_t)b * T_SEQ * D_MODEL + h * HEAD_DIM;
  {
    const int qr = tid >> 2, qc = (tid & 3) * 16;  // 64x64, 16 h/thr
    copy16_async(&Qp[(size_t)(q0 + qr) * D_MODEL + qc], &Qs[qr][qc]);
    copy16_async(&Qp[(size_t)(q0 + qr) * D_MODEL + qc + 8], &Qs[qr][qc + 8]);
    const int kr = tid >> 1, kc = (tid & 1) * 32;  // 128x64, 32 h/thr
#pragma unroll
    for (int j = 0; j < 32; j += 8)
      copy16_async(&Kp[(size_t)(p0 + kr) * D_MODEL + kc + j], &Ks[kr][kc + j]);
  }
  async_wait0();
  __syncthreads();

  const int mB = (wave & 1) * 2;   // query subtile pair
  const int nB = (wave >> 1) * 2;  // key subtile pair (0..7)
  v8f acc00 = {}, acc01 = {}, acc10 = {}, acc11 = {};
#pragma unroll
  for (int ks = 0; ks < HEAD_DIM; ks += 32) {
    v16h a0 = frag_a(&Qs[(mB + 0) * 16][ks], 72);
    v16h a1 = frag_a(&Qs[(mB + 1) * 16][ks], 72);
    v16h b0 = frag_bT(&Ks[(nB + 0) * 16][ks], 72);
    v16h b1 = frag_bT(&Ks[(nB + 1) * 16][ks], 72);
    acc00 = wmma32(a0, b0, acc00);
    acc01 = wmma32(a0, b1, acc01);
    acc10 = wmma32(a1, b0, acc10);
    acc11 = wmma32(a1, b1, acc11);
  }

  const float* mp = mask + (size_t)b * T_SEQ * T_SEQ;  // [B,1,T,T]
  float* op = attnW + (size_t)bh * T_SEQ * T_SEQ;
  const int gk0 = p0 + (nB + 0) * 16 + col;
  const int gk1 = p0 + (nB + 1) * 16 + col;
  // phase 1: batch all mask loads (avoids load->wait->store serialization)
  float mv[4][8];
#pragma unroll
  for (int r = 0; r < 8; ++r) {
    const int gq0 = q0 + (mB + 0) * 16 + r + 8 * hi;
    const int gq1 = q0 + (mB + 1) * 16 + r + 8 * hi;
    mv[0][r] = mp[(size_t)gq0 * T_SEQ + gk0];
    mv[1][r] = mp[(size_t)gq0 * T_SEQ + gk1];
    mv[2][r] = mp[(size_t)gq1 * T_SEQ + gk0];
    mv[3][r] = mp[(size_t)gq1 * T_SEQ + gk1];
  }
  // phase 2: add + store
#pragma unroll
  for (int r = 0; r < 8; ++r) {
    const int gq0 = q0 + (mB + 0) * 16 + r + 8 * hi;
    const int gq1 = q0 + (mB + 1) * 16 + r + 8 * hi;
    op[(size_t)gq0 * T_SEQ + gk0] = acc00[r] + mv[0][r];
    op[(size_t)gq0 * T_SEQ + gk1] = acc01[r] + mv[1][r];
    op[(size_t)gq1 * T_SEQ + gk0] = acc10[r] + mv[2][r];
    op[(size_t)gq1 * T_SEQ + gk1] = acc11[r] + mv[3][r];
  }
}

// ---------------------------------------------------------------------------
// Stage 4: row-wise softmax in place (block of 256 per row of 2048)
// ---------------------------------------------------------------------------
__global__ __launch_bounds__(256) void softmax_kernel(float* __restrict__ attnW) {
  float* p = attnW + (size_t)blockIdx.x * T_SEQ;
  const int tid = threadIdx.x;
  const int wave = tid >> 5, lane = tid & 31;
  __shared__ float smax[8];
  __shared__ float ssum[8];
  float v[8];
  float mx = -3.4e38f;
#pragma unroll
  for (int i = 0; i < 8; ++i) {
    v[i] = p[tid + i * 256];
    mx = fmaxf(mx, v[i]);
  }
#pragma unroll
  for (int o = 16; o > 0; o >>= 1) mx = fmaxf(mx, __shfl_xor(mx, o, 32));
  if (lane == 0) smax[wave] = mx;
  __syncthreads();
#pragma unroll
  for (int i = 0; i < 8; ++i) mx = fmaxf(mx, smax[i]);
  float s = 0.f;
#pragma unroll
  for (int i = 0; i < 8; ++i) {
    v[i] = __expf(v[i] - mx);
    s += v[i];
  }
#pragma unroll
  for (int o = 16; o > 0; o >>= 1) s += __shfl_xor(s, o, 32);
  if (lane == 0) ssum[wave] = s;
  __syncthreads();
  s = 0.f;
#pragma unroll
  for (int i = 0; i < 8; ++i) s += ssum[i];
  const float inv = 1.0f / s;
#pragma unroll
  for (int i = 0; i < 8; ++i) p[tid + i * 256] = v[i] * inv;
}

// ---------------------------------------------------------------------------
// Stage 5: ctx = P . V ; block tile 128q x 64d, K-step 64, 2x2 subtiles/wave.
// P converted fp32->f16 on stage; V staged async. Output f16 in [B,T,D].
// ---------------------------------------------------------------------------
__global__ __launch_bounds__(256) void ctx_kernel(
    const float* __restrict__ attnW, const _Float16* __restrict__ V,
    _Float16* __restrict__ ctx) {
  __shared__ _Float16 Ps[128][72];
  __shared__ _Float16 Vs[64][72];
  const int tid  = threadIdx.x;
  const int wave = tid >> 5, lane = tid & 31;
  const int hi = lane >> 4, col = lane & 15;
  const int bh = blockIdx.y, b = bh >> 4, h = bh & 15;
  const int q0 = blockIdx.x * 128;
  const float* Pp = attnW + (size_t)bh * T_SEQ * T_SEQ;
  const _Float16* Vp = V + (size_t)b * T_SEQ * D_MODEL + h * HEAD_DIM;
  const int pr = tid >> 1, pc = (tid & 1) * 32;  // P loader: 128x64, 32/thr
  const int vr = tid >> 2, vc = (tid & 3) * 16;  // V loader: 64x64, 16/thr
  const int mB = (wave >> 1) * 2;  // q subtile pair (0,2,4,6)
  const int nB = (wave & 1) * 2;   // d subtile pair (0,2)
  v8f acc00 = {}, acc01 = {}, acc10 = {}, acc11 = {};

  for (int k0 = 0; k0 < T_SEQ; k0 += 64) {
    copy16_async(&Vp[(size_t)(k0 + vr) * D_MODEL + vc], &Vs[vr][vc]);
    copy16_async(&Vp[(size_t)(k0 + vr) * D_MODEL + vc + 8], &Vs[vr][vc + 8]);
#pragma unroll
    for (int j = 0; j < 32; j += 4) {
      float4 pv = *(const float4*)&Pp[(size_t)(q0 + pr) * T_SEQ + k0 + pc + j];
      Ps[pr][pc + j]     = (_Float16)pv.x;
      Ps[pr][pc + j + 1] = (_Float16)pv.y;
      Ps[pr][pc + j + 2] = (_Float16)pv.z;
      Ps[pr][pc + j + 3] = (_Float16)pv.w;
    }
    async_wait0();
    __syncthreads();
#pragma unroll
    for (int ks = 0; ks < 64; ks += 32) {
      v16h a0 = frag_a(&Ps[(mB + 0) * 16][ks], 72);
      v16h a1 = frag_a(&Ps[(mB + 1) * 16][ks], 72);
      v16h b0 = frag_b(&Vs[ks][(nB + 0) * 16], 72);
      v16h b1 = frag_b(&Vs[ks][(nB + 1) * 16], 72);
      acc00 = wmma32(a0, b0, acc00);
      acc01 = wmma32(a0, b1, acc01);
      acc10 = wmma32(a1, b0, acc10);
      acc11 = wmma32(a1, b1, acc11);
    }
    __syncthreads();
  }

#pragma unroll
  for (int r = 0; r < 8; ++r) {
    const int gq0 = q0 + (mB + 0) * 16 + r + 8 * hi;
    const int gq1 = q0 + (mB + 1) * 16 + r + 8 * hi;
    const int gd0 = (nB + 0) * 16 + col;
    const int gd1 = (nB + 1) * 16 + col;
    _Float16* c0 = &ctx[((size_t)b * T_SEQ + gq0) * D_MODEL + h * HEAD_DIM];
    _Float16* c1 = &ctx[((size_t)b * T_SEQ + gq1) * D_MODEL + h * HEAD_DIM];
    c0[gd0] = (_Float16)acc00[r];
    c0[gd1] = (_Float16)acc01[r];
    c1[gd0] = (_Float16)acc10[r];
    c1[gd1] = (_Float16)acc11[r];
  }
}

// ---------------------------------------------------------------------------
// Host launcher
// ---------------------------------------------------------------------------
extern "C" void kernel_launch(void* const* d_in, const int* in_sizes, int n_in,
                              void* d_out, int out_size, void* d_ws,
                              size_t ws_size, hipStream_t stream) {
  (void)in_sizes; (void)n_in; (void)out_size; (void)ws_size;
  const float* hs   = (const float*)d_in[0];
  const float* mask = (const float*)d_in[1];
  const float* Wq   = (const float*)d_in[2];
  const float* bq   = (const float*)d_in[3];
  const float* Wk   = (const float*)d_in[4];
  const float* bk   = (const float*)d_in[5];
  const float* Wv   = (const float*)d_in[6];
  const float* bv   = (const float*)d_in[7];
  const float* Wo   = (const float*)d_in[8];
  const float* bo   = (const float*)d_in[9];
  // d_in[10] = num_heads (device scalar) — fixed at 16 per reference.

  float* attn_out = (float*)d_out;                                // [B,T,D]
  float* attn_w = attn_out + (size_t)N_BATCH * T_SEQ * D_MODEL;   // [B,H,T,T]

  const size_t nX = (size_t)N_BATCH * T_SEQ * D_MODEL;  // 4,194,304
  const size_t nW = (size_t)D_MODEL * D_MODEL;          // 1,048,576

  _Float16* Xh  = (_Float16*)d_ws;
  _Float16* Wqh = Xh + nX;
  _Float16* Wkh = Wqh + nW;
  _Float16* Wvh = Wkh + nW;
  _Float16* Woh = Wvh + nW;
  _Float16* Qh  = Woh + nW;
  _Float16* Kh  = Qh + nX;
  _Float16* Vh  = Kh + nX;
  _Float16* Ch  = Vh + nX;  // total ~48 MB of f16 workspace

  cvt_f32_to_f16<<<(int)(nX / 1024), 256, 0, stream>>>(hs, Xh, (int)nX);
  cvt_f32_to_f16<<<(int)(nW / 1024), 256, 0, stream>>>(Wq, Wqh, (int)nW);
  cvt_f32_to_f16<<<(int)(nW / 1024), 256, 0, stream>>>(Wk, Wkh, (int)nW);
  cvt_f32_to_f16<<<(int)(nW / 1024), 256, 0, stream>>>(Wv, Wvh, (int)nW);
  cvt_f32_to_f16<<<(int)(nW / 1024), 256, 0, stream>>>(Wo, Woh, (int)nW);

  dim3 gg(64, 8);  // M=4096/64, N=1024/128
  // Q = (X*Wq + bq) * hd^-0.5 ; K,V unscaled
  gemm_bias<false><<<gg, 256, 0, stream>>>(Xh, Wqh, bq, 0.125f, Qh, nullptr,
                                           4096, 1024, 1024);
  gemm_bias<false><<<gg, 256, 0, stream>>>(Xh, Wkh, bk, 1.0f, Kh, nullptr,
                                           4096, 1024, 1024);
  gemm_bias<false><<<gg, 256, 0, stream>>>(Xh, Wvh, bv, 1.0f, Vh, nullptr,
                                           4096, 1024, 1024);

  scores_kernel<<<dim3(32, 16, 32), 256, 0, stream>>>(Qh, Kh, mask, attn_w);
  softmax_kernel<<<dim3(65536), 256, 0, stream>>>(attn_w);
  ctx_kernel<<<dim3(16, 32), 256, 0, stream>>>(attn_w, Vh, Ch);

  gemm_bias<true><<<gg, 256, 0, stream>>>(Ch, Woh, bo, 1.0f, nullptr, attn_out,
                                          4096, 1024, 1024);
}
